// UNet_43258910605941
// MI455X (gfx1250) — matrix operand order repaired and checked by reading, not compiled, not run on hardware
//
#include <hip/hip_runtime.h>
#include <hip/hip_bf16.h>
#include <algorithm>

// ---------------------------------------------------------------------------
// Types for CDNA5 WMMA (wave32, v_wmma_f32_16x16x32_bf16)
// ---------------------------------------------------------------------------
typedef __bf16 bf16_t;
typedef bf16_t v16bf __attribute__((ext_vector_type(16)));
typedef bf16_t v8bf  __attribute__((ext_vector_type(8)));
typedef float  v8f   __attribute__((ext_vector_type(8)));

// ---------------------------------------------------------------------------
// Input image fp32 [S] -> bf16 NDHWC [S][32] (channel 0 real, 1..31 zero pad)
// ---------------------------------------------------------------------------
__global__ void unet_to_bf16_pad(const float* __restrict__ x, bf16_t* __restrict__ y,
                                 int S) {
    int i = blockIdx.x * blockDim.x + threadIdx.x;
    if (i >= S * 32) return;
    int s = i >> 5, c = i & 31;
    y[i] = (bf16_t)((c == 0) ? x[s] : 0.f);
}

// ---------------------------------------------------------------------------
// Pack conv weights (Cout, Cin, 3,3,3) fp32 into the exact per-lane B-fragment
// register image, K ordered as k = tap*Cinp + ci (zero-padded ci >= Cin):
//   blob[(((ct*nchunk + ch)*32 + lane)*16 + e]  with
//   lane = hi*16 + n,  k = ch*32 + hi*16 + e,  n = o & 15, ct = o >> 4
// One contiguous 32-byte load per lane per chunk reconstructs the fragment.
// ---------------------------------------------------------------------------
__global__ void unet_pack_w(const float* __restrict__ w, bf16_t* __restrict__ wp,
                            int Cout, int Cin, int Cinp) {
    int Kp = 27 * Cinp;
    int total = Cout * Kp;
    int idx = blockIdx.x * blockDim.x + threadIdx.x;
    if (idx >= total) return;
    int o = idx / Kp, k = idx - o * Kp;
    int tap = k / Cinp, ci = k - tap * Cinp;
    float v = (ci < Cin) ? w[((size_t)o * Cin + ci) * 27 + tap] : 0.f;
    int ct = o >> 4, n = o & 15;
    int ch = k >> 5, r = k & 31, hi = r >> 4, e = r & 15;
    int nchunk = Kp >> 5;
    wp[(((size_t)ct * nchunk + ch) * 32 + (hi * 16 + n)) * 16 + e] = (bf16_t)v;
}

// ---------------------------------------------------------------------------
// Implicit-GEMM conv3x3x3 (pad=1), NDHWC, via v_wmma_f32_16x16x32_bf16.
// One wave computes a 32-voxel x 16-outchannel tile as TWO 16x16 D tiles that
// share every B fragment (2 independent WMMA chains per K-chunk).
//   in  : bf16 [S][Cinp]   (Cinp % 32 == 0, pad channels are zero)
//   wb  : pre-shuffled blob (see unet_pack_w)
//   zpad: >= Cinp bf16 zeros; out-of-bounds taps read from here (no exec masks)
//   out : fp32 [S][Cout]
// ---------------------------------------------------------------------------
__global__ __launch_bounds__(32)
void unet_conv3_wmma(const bf16_t* __restrict__ in, const bf16_t* __restrict__ wb,
                     const bf16_t* __restrict__ zpad, float* __restrict__ out,
                     int Cinp, int Cout, int D, int H, int W) {
    const int tile = blockIdx.x;     // 32-voxel tile
    const int ct   = blockIdx.y;     // 16-outchannel tile
    const int lane = threadIdx.x;    // 0..31 (wave32)
    const int hi   = lane >> 4;
    const int ln   = lane & 15;

    const int s0  = tile * 32;
    const int sA0 = s0 + ln;         // A rows for D-tile 0
    const int sA1 = s0 + 16 + ln;    // A rows for D-tile 1
    const int x0 = sA0 % W, y0 = (sA0 / W) % H, z0 = sA0 / (W * H);
    const int x1 = sA1 % W, y1 = (sA1 / W) % H, z1 = sA1 / (W * H);

    const int nchunk_c = Cinp >> 5;  // K-chunks per tap
    const v16bf* __restrict__ wptr =
        (const v16bf*)wb + (size_t)ct * (27 * nchunk_c) * 32 + lane;

    v8f acc0 = {0.f, 0.f, 0.f, 0.f, 0.f, 0.f, 0.f, 0.f};
    v8f acc1 = {0.f, 0.f, 0.f, 0.f, 0.f, 0.f, 0.f, 0.f};

    for (int dz = -1; dz <= 1; ++dz) {
        for (int dy = -1; dy <= 1; ++dy) {
            for (int dx = -1; dx <= 1; ++dx) {
                const int za = z0 + dz, ya = y0 + dy, xa = x0 + dx;
                const int zb = z1 + dz, yb = y1 + dy, xb = x1 + dx;
                const bool ia = (unsigned)za < (unsigned)D &&
                                (unsigned)ya < (unsigned)H &&
                                (unsigned)xa < (unsigned)W;
                const bool ib = (unsigned)zb < (unsigned)D &&
                                (unsigned)yb < (unsigned)H &&
                                (unsigned)xb < (unsigned)W;
                // Zero-page fallback: no exec masking in the K loop.
                const v8bf* pa = (const v8bf*)
                    (ia ? in + (size_t)((za * H + ya) * W + xa) * Cinp : zpad);
                const v8bf* pb = (const v8bf*)
                    (ib ? in + (size_t)((zb * H + yb) * W + xb) * Cinp : zpad);

                // One speculative prefetch of next tap's weight stream.
                __builtin_prefetch(wptr + (size_t)32 * nchunk_c, 0, 1);

                for (int cb = 0; cb < nchunk_c; ++cb) {
                    v16bf b = *wptr;
                    wptr += 32;
                    v8bf a0l = pa[4 * cb + 2 * hi];       // ch hi*16..+7
                    v8bf a0h = pa[4 * cb + 2 * hi + 1];   // ch hi*16+8..+7
                    v8bf a1l = pb[4 * cb + 2 * hi];
                    v8bf a1h = pb[4 * cb + 2 * hi + 1];
                    v16bf a0 = __builtin_shufflevector(
                        a0l, a0h, 0, 1, 2, 3, 4, 5, 6, 7, 8, 9, 10, 11, 12, 13, 14, 15);
                    v16bf a1 = __builtin_shufflevector(
                        a1l, a1h, 0, 1, 2, 3, 4, 5, 6, 7, 8, 9, 10, 11, 12, 13, 14, 15);
                    acc0 = __builtin_amdgcn_wmma_f32_16x16x32_bf16(
                        false, a0, false, b, (short)0, acc0, false, false);
                    acc1 = __builtin_amdgcn_wmma_f32_16x16x32_bf16(
                        false, a1, false, b, (short)0, acc1, false, false);
                }
            }
        }
    }

    // C/D layout: lane ln = column N (channel); VGPR r = row M = r + 8*hi.
    const int c = ct * 16 + ln;
    #pragma unroll
    for (int r = 0; r < 8; ++r)
        out[(size_t)(s0 + r + 8 * hi) * Cout + c] = acc0[r];
    #pragma unroll
    for (int r = 0; r < 8; ++r)
        out[(size_t)(s0 + 16 + r + 8 * hi) * Cout + c] = acc1[r];
}

// ---------------------------------------------------------------------------
// BatchNorm stats over fp32 [S][C]: per-channel sum / sum^2
// ---------------------------------------------------------------------------
__global__ void unet_zero_f32(float* p, int n) {
    int i = blockIdx.x * blockDim.x + threadIdx.x;
    if (i < n) p[i] = 0.f;
}

__global__ __launch_bounds__(256)
void unet_bn_stats(const float* __restrict__ x, float* __restrict__ stats, int C, int S) {
    __shared__ float s1[256], s2[256];
    int c = blockIdx.y;
    float a = 0.f, b = 0.f;
    for (int i = blockIdx.x * 256 + threadIdx.x; i < S; i += gridDim.x * 256) {
        float v = x[(size_t)i * C + c];
        a += v;
        b += v * v;
    }
    s1[threadIdx.x] = a;
    s2[threadIdx.x] = b;
    __syncthreads();
    for (int o = 128; o > 0; o >>= 1) {
        if (threadIdx.x < o) {
            s1[threadIdx.x] += s1[threadIdx.x + o];
            s2[threadIdx.x] += s2[threadIdx.x + o];
        }
        __syncthreads();
    }
    if (threadIdx.x == 0) {
        atomicAdd(&stats[c], s1[0]);
        atomicAdd(&stats[C + c], s2[0]);
    }
}

// ---------------------------------------------------------------------------
// Normalize + LeakyReLU -> bf16 NDHWC, writing channels [coff, Ct) of a buffer
// with row stride Ct. Channels >= C (real) are written as zero (K padding).
// ---------------------------------------------------------------------------
__global__ void unet_bn_lrelu(const float* __restrict__ x, const float* __restrict__ stats,
                              bf16_t* __restrict__ y, int C, int S, int Ct, int coff) {
    int nc = Ct - coff;
    int i = blockIdx.x * blockDim.x + threadIdx.x;
    if (i >= S * nc) return;
    int s = i / nc, c = i - s * nc;
    float v = 0.f;
    if (c < C) {
        float inv = 1.f / (float)S;              // batch = 1
        float m   = stats[c] * inv;
        float var = stats[C + c] * inv - m * m;
        v = (x[(size_t)s * C + c] - m) * rsqrtf(var + 1e-5f);
        v = (v >= 0.f) ? v : 0.01f * v;
    }
    y[(size_t)s * Ct + coff + c] = (bf16_t)v;
}

// ---------------------------------------------------------------------------
// 2x2x2 max pool, NDHWC (pad channels are 0 in, 0 out)
// ---------------------------------------------------------------------------
__global__ void unet_maxpool2(const bf16_t* __restrict__ in, bf16_t* __restrict__ out,
                              int Cp, int D, int H, int W) {
    int Do = D >> 1, Ho = H >> 1, Wo = W >> 1;
    int So = Do * Ho * Wo;
    int i = blockIdx.x * blockDim.x + threadIdx.x;
    if (i >= So * Cp) return;
    int s = i / Cp, c = i - s * Cp;
    int x = s % Wo, y = (s / Wo) % Ho, z = s / (Wo * Ho);
    float m = -3.0e38f;
    #pragma unroll
    for (int dz = 0; dz < 2; ++dz)
        #pragma unroll
        for (int dy = 0; dy < 2; ++dy)
            #pragma unroll
            for (int dx = 0; dx < 2; ++dx) {
                size_t si = (size_t)(((2 * z + dz) * H + 2 * y + dy) * W + 2 * x + dx);
                m = fmaxf(m, (float)in[si * Cp + c]);
            }
    out[i] = (bf16_t)m;
}

// ---------------------------------------------------------------------------
// Skip-concat gather: cat[s*Ct + c] = skip[s*Cps + c], c < Cskip
// ---------------------------------------------------------------------------
__global__ void unet_copy_skip(const bf16_t* __restrict__ skip, bf16_t* __restrict__ cat,
                               int Cskip, int Cps, int Ct, int S) {
    int i = blockIdx.x * blockDim.x + threadIdx.x;
    if (i >= S * Cskip) return;
    int s = i / Cskip, c = i - s * Cskip;
    cat[(size_t)s * Ct + c] = skip[(size_t)s * Cps + c];
}

// ---------------------------------------------------------------------------
// "ConvTranspose" = nearest upsample x2 + circular 2x2x2 conv (== FFT path).
// NDHWC: in bf16 [Si][Cin], out fp32 [So][Cout]. Dims are powers of two.
// ---------------------------------------------------------------------------
__global__ void unet_upconv(const bf16_t* __restrict__ in, const float* __restrict__ w,
                            float* __restrict__ out, int Cin, int Cout,
                            int Di, int Hi, int Wi) {
    int Do = 2 * Di, Ho = 2 * Hi, Wo = 2 * Wi;
    int So = Do * Ho * Wo;
    int i = blockIdx.x * blockDim.x + threadIdx.x;
    if (i >= So * Cout) return;
    int s = i / Cout, o = i - s * Cout;
    int x = s % Wo, y = (s / Wo) % Ho, z = s / (Wo * Ho);
    float acc = 0.f;
    #pragma unroll
    for (int dz = 0; dz < 2; ++dz)
        #pragma unroll
        for (int dy = 0; dy < 2; ++dy)
            #pragma unroll
            for (int dx = 0; dx < 2; ++dx) {
                int zz = (z - dz) & (Do - 1);
                int yy = (y - dy) & (Ho - 1);
                int xx = (x - dx) & (Wo - 1);
                const bf16_t* pin =
                    in + (size_t)(((zz >> 1) * Hi + (yy >> 1)) * Wi + (xx >> 1)) * Cin;
                const float* pw = w + ((size_t)o * Cin) * 8 + ((dz * 2 + dy) * 2 + dx);
                for (int ci = 0; ci < Cin; ++ci)
                    acc += pw[(size_t)ci * 8] * (float)pin[ci];
            }
    out[i] = acc;
}

// ---------------------------------------------------------------------------
// Final 1x1x1 conv 16 -> 1 + bias; a3 is NDHWC [S][32] (16 real channels)
// ---------------------------------------------------------------------------
__global__ void unet_final_conv(const bf16_t* __restrict__ a, const float* __restrict__ w,
                                const float* __restrict__ bias, float* __restrict__ out,
                                int S) {
    int s = blockIdx.x * blockDim.x + threadIdx.x;
    if (s >= S) return;
    float acc = bias[0];
    #pragma unroll
    for (int c = 0; c < 16; ++c) acc += w[c] * (float)a[(size_t)s * 32 + c];
    out[s] = acc;
}

// ---------------------------------------------------------------------------
// Host orchestration
// ---------------------------------------------------------------------------
extern "C" void kernel_launch(void* const* d_in, const int* in_sizes, int n_in,
                              void* d_out, int out_size, void* d_ws, size_t ws_size,
                              hipStream_t stream) {
    (void)in_sizes; (void)n_in; (void)out_size; (void)ws_size;

    const float* x       = (const float*)d_in[0];
    const float* w_final = (const float*)d_in[18];
    const float* b_final = (const float*)d_in[19];
    float*       out     = (float*)d_out;

    const int S64 = 64 * 64 * 64, S32 = 32 * 32 * 32, S16 = 16 * 16 * 16, S8 = 8 * 8 * 8;

    // ---- workspace arena -------------------------------------------------
    char*  base = (char*)d_ws;
    size_t off  = 0;
    auto alloc = [&](size_t bytes) -> void* {
        void* p = base + off;
        off = (off + bytes + 255) & ~(size_t)255;
        return p;
    };

    bf16_t* xb   = (bf16_t*)alloc((size_t)S64 * 32 * 2);  // input, padded to 32ch
    float*  tmp  = (float*) alloc((size_t)S64 * 16 * 4);  // largest fp32 conv temp
    bf16_t* tact = (bf16_t*)alloc((size_t)S64 * 32 * 2);  // in-block activation temp
    bf16_t* x1   = (bf16_t*)alloc((size_t)S64 * 32 * 2);
    bf16_t* p1   = (bf16_t*)alloc((size_t)S32 * 32 * 2);
    bf16_t* x2   = (bf16_t*)alloc((size_t)S32 * 32 * 2);
    bf16_t* p2   = (bf16_t*)alloc((size_t)S16 * 32 * 2);
    bf16_t* x3   = (bf16_t*)alloc((size_t)S16 * 64 * 2);
    bf16_t* p3   = (bf16_t*)alloc((size_t)S8  * 64 * 2);
    bf16_t* x4   = (bf16_t*)alloc((size_t)S8  * 128 * 2);
    bf16_t* c1   = (bf16_t*)alloc((size_t)S16 * 128 * 2);
    bf16_t* a1   = (bf16_t*)alloc((size_t)S16 * 64 * 2);
    bf16_t* c2   = (bf16_t*)alloc((size_t)S32 * 64 * 2);
    bf16_t* a2   = (bf16_t*)alloc((size_t)S32 * 32 * 2);
    bf16_t* c3   = (bf16_t*)alloc((size_t)S64 * 32 * 2);
    bf16_t* a3   = (bf16_t*)alloc((size_t)S64 * 32 * 2);
    float*  stats = (float*)alloc(512 * 4);
    bf16_t* zpad = (bf16_t*)alloc(512);                   // zero page for OOB taps

    // ---- pack 3x3x3 weights into the B-fragment lane image ---------------
    struct PW { int idx, Cout, Cin, Cinp; };
    const PW specs[14] = {
        {1, 16, 1, 32},    {2, 16, 16, 32},  {3, 32, 16, 32},  {4, 32, 32, 32},
        {5, 64, 32, 32},   {6, 64, 64, 64},  {7, 128, 64, 64}, {8, 128, 128, 128},
        {10, 64, 128, 128},{11, 64, 64, 64}, {13, 32, 64, 64}, {14, 32, 32, 32},
        {16, 16, 32, 32},  {17, 16, 16, 32}
    };
    bf16_t* wpk[20] = {nullptr};
    for (int i = 0; i < 14; ++i) {
        const PW& s = specs[i];
        int Kp = 27 * s.Cinp;
        bf16_t* dst = (bf16_t*)alloc((size_t)s.Cout * Kp * 2);
        wpk[s.idx] = dst;
        int total = s.Cout * Kp;
        unet_pack_w<<<(total + 255) / 256, 256, 0, stream>>>(
            (const float*)d_in[s.idx], dst, s.Cout, s.Cin, s.Cinp);
    }
    unet_zero_f32<<<1, 128, 0, stream>>>((float*)zpad, 128);   // zero the zero page

    // ---- per-layer helpers ----------------------------------------------
    auto bn_pass = [&](float* conv, bf16_t* act, int Cout, int S, int Ct, int coff) {
        unet_zero_f32<<<1, 256, 0, stream>>>(stats, 2 * Cout);
        int nblk = std::min((S + 255) / 256, 128);
        unet_bn_stats<<<dim3(nblk, Cout), 256, 0, stream>>>(conv, stats, Cout, S);
        int n = S * (Ct - coff);
        unet_bn_lrelu<<<(n + 255) / 256, 256, 0, stream>>>(conv, stats, act,
                                                           Cout, S, Ct, coff);
    };

    auto run_conv = [&](const bf16_t* in, int inCp, int widx, int Cout, int d,
                        bf16_t* act_out, int outCt) {
        int S = d * d * d;
        dim3 grid(S / 32, Cout / 16);
        unet_conv3_wmma<<<grid, 32, 0, stream>>>(in, wpk[widx], zpad, tmp, inCp, Cout,
                                                 d, d, d);
        bn_pass(tmp, act_out, Cout, S, outCt, 0);
    };

    auto run_pool = [&](const bf16_t* in, bf16_t* outb, int Cp, int d) {
        int n = (d / 2) * (d / 2) * (d / 2) * Cp;
        unet_maxpool2<<<(n + 255) / 256, 256, 0, stream>>>(in, outb, Cp, d, d, d);
    };

    auto run_up = [&](const bf16_t* a_in, const float* wt, int Cin, int Cout, int di,
                      const bf16_t* skip, int skipCp, int Cskip, bf16_t* cat, int Ct) {
        int d = 2 * di, S = d * d * d;
        int n = S * Cout;
        unet_upconv<<<(n + 255) / 256, 256, 0, stream>>>(a_in, wt, tmp, Cin, Cout,
                                                         di, di, di);
        bn_pass(tmp, cat, Cout, S, Ct, Cskip);
        int nc = S * Cskip;
        unet_copy_skip<<<(nc + 255) / 256, 256, 0, stream>>>(skip, cat, Cskip,
                                                             skipCp, Ct, S);
    };

    // ---- forward ---------------------------------------------------------
    unet_to_bf16_pad<<<(S64 * 32 + 255) / 256, 256, 0, stream>>>(x, xb, S64);

    // encoder
    run_conv(xb,   32, 1, 16, 64, tact, 32);
    run_conv(tact, 32, 2, 16, 64, x1, 32);
    run_pool(x1, p1, 32, 64);
    run_conv(p1,   32, 3, 32, 32, tact, 32);
    run_conv(tact, 32, 4, 32, 32, x2, 32);
    run_pool(x2, p2, 32, 32);
    run_conv(p2,   32, 5, 64, 16, tact, 64);
    run_conv(tact, 64, 6, 64, 16, x3, 64);
    run_pool(x3, p3, 64, 16);
    run_conv(p3,   64, 7, 128, 8, tact, 128);
    run_conv(tact, 128, 8, 128, 8, x4, 128);

    // decoder
    run_up(x4, (const float*)d_in[9], 128, 64, 8, x3, 64, 64, c1, 128);
    run_conv(c1,   128, 10, 64, 16, tact, 64);
    run_conv(tact, 64,  11, 64, 16, a1, 64);

    run_up(a1, (const float*)d_in[12], 64, 32, 16, x2, 32, 32, c2, 64);
    run_conv(c2,   64, 13, 32, 32, tact, 32);
    run_conv(tact, 32, 14, 32, 32, a2, 32);

    run_up(a2, (const float*)d_in[15], 32, 16, 32, x1, 32, 16, c3, 32);
    run_conv(c3,   32, 16, 16, 64, tact, 32);
    run_conv(tact, 32, 17, 16, 64, a3, 32);

    unet_final_conv<<<(S64 + 255) / 256, 256, 0, stream>>>(a3, w_final, b_final,
                                                           out, S64);
}
